// MDMTRec_31894427140173
// MI455X (gfx1250) — matrix operand order repaired and checked by reading, not compiled, not run on hardware
//
#include <hip/hip_runtime.h>
#include <hip/hip_bf16.h>

// ---------------------------------------------------------------- types
typedef __attribute__((ext_vector_type(16))) __bf16        v16bf;
typedef __attribute__((ext_vector_type(8)))  __bf16        v8bf;
typedef __attribute__((ext_vector_type(8)))  float         v8f;
typedef __attribute__((ext_vector_type(4)))  unsigned int  v4u;
typedef __attribute__((ext_vector_type(8)))  int           v8i;
typedef __attribute__((ext_vector_type(4)))  int           v4i;

static constexpr int BB   = 16384;   // batch
static constexpr int FF   = 32;      // features
static constexpr int VV   = 50000;   // vocab
static constexpr int EMBD = 16;      // emb dim
static constexpr int IN   = 512;     // FF*EMBD
static constexpr int KC   = 128;     // K-chunk staged in LDS (16 KB panel)

__device__ __forceinline__ unsigned short f2bf(float f) {
    unsigned int u = __builtin_bit_cast(unsigned int, f);
    unsigned int r = (u + 0x7FFFu + ((u >> 16) & 1u)) >> 16;
    return (unsigned short)r;
}

__device__ __forceinline__ v16bf load_frag(const unsigned short* p0,
                                           const unsigned short* p1) {
    v4u lo = *(const v4u*)p0;      // 8 bf16
    v4u hi = *(const v4u*)p1;      // 8 bf16
    v8bf l = __builtin_bit_cast(v8bf, lo);
    v8bf h = __builtin_bit_cast(v8bf, hi);
    return __builtin_shufflevector(l, h, 0,1,2,3,4,5,6,7,8,9,10,11,12,13,14,15);
}

__device__ __forceinline__ v8f wmma_bf16(v16bf a, v16bf b, v8f c) {
    return __builtin_amdgcn_wmma_f32_16x16x32_bf16(false, a, false, b,
                                                   (short)0, c, false, false);
}

// ---- Tensor Data Mover: stage a [64 x kcs] bf16 weight panel into LDS.
// D# per CDNA5 ISA ch.8: group0 = {count/type, lds_addr, global_addr},
// group1 = {data_size, tensor dims, tile dims, dim0 stride}.
__device__ __forceinline__ void tdm_load_b_panel(const unsigned short* gsrc,
                                                 unsigned ldsAddr,
                                                 int K, int kcs) {
    unsigned long long ga = (unsigned long long)(uintptr_t)gsrc;
    v4u g0;
    g0.x = 1u;                                               // count=1 (valid)
    g0.y = ldsAddr;                                          // lds_addr
    g0.z = (unsigned)(ga & 0xFFFFFFFFu);                     // global_addr[31:0]
    g0.w = (unsigned)((ga >> 32) & 0x1FFFFFFu) | (2u << 30); // addr[56:32]|type=2
    v8i g1;
    g1[0] = (1 << 16);                          // data_size=1 => 2 bytes
    g1[1] = (K & 0xFFFF) << 16;                 // tensor_dim0[15:0]
    g1[2] = ((K >> 16) & 0xFFFF) | (64 << 16);  // tensor_dim0[31:16]|tensor_dim1=64
    g1[3] = (kcs & 0xFFFF) << 16;               // tile_dim0 = kcs
    g1[4] = 64;                                 // tile_dim1=64, tile_dim2=0
    g1[5] = K;                                  // tensor_dim0_stride[31:0]
    g1[6] = 0;
    g1[7] = 0;
    v4i z4 = {0, 0, 0, 0};
#if __clang_major__ >= 23
    v8i z8 = {0, 0, 0, 0, 0, 0, 0, 0};
    __builtin_amdgcn_tensor_load_to_lds(g0, g1, z4, z4, z8, 0);
#else
    __builtin_amdgcn_tensor_load_to_lds(g0, g1, z4, z4, 0);
#endif
}

// ------------------------------------------------- generic bf16 WMMA GEMM
// C[M,N] = A[M,K](bf16 row-major) @ Wt[N,K](bf16 pre-transposed) + bias.
// Block = 4 waves, each wave owns a 32x64 strip (8 accumulators, 8 WMMAs
// per 32-K step). The shared 64-row B panel is DMA'd into LDS by the TDM
// once per K-chunk and read back with ds_load_b128 by all 4 waves.
__global__ void __launch_bounds__(128)
gemm_bf16_kernel(const unsigned short* __restrict__ A,
                 const unsigned short* __restrict__ Wt,
                 const float* __restrict__ bias,
                 float* __restrict__ C,
                 int M, int N, int K,
                 const int* __restrict__ dom, int domSel) {
    __shared__ unsigned short btile[64 * KC];   // 16 KB weight panel

    const int lane = threadIdx.x & 31;
    const int wave = threadIdx.x >> 5;
    const int row0 = blockIdx.x * 128 + wave * 32;
    const int col0 = blockIdx.y * 64;
    const int mrow = lane & 15;
    const int c8   = (lane >> 4) * 8;    // A K-offset within 32-block
    const int kb   = (lane >> 4) * 16;   // unused; B fragments also use c8

    v8f zero = {0,0,0,0,0,0,0,0};
    v8f acc[8];
#pragma unroll
    for (int i = 0; i < 8; ++i) acc[i] = zero;

    const unsigned short* aRow0 = A + (size_t)(row0 + mrow) * K;
    const unsigned short* aRow1 = aRow0 + (size_t)16 * K;
    const int kcs = (K < KC) ? K : KC;

    for (int k0 = 0; k0 < K; k0 += kcs) {
        if (wave == 0) {
            tdm_load_b_panel(Wt + (size_t)col0 * K + k0,
                             (unsigned)(uintptr_t)btile, K, kcs);
            __builtin_amdgcn_s_wait_tensorcnt(0);
        }
        __syncthreads();                         // panel visible to all waves
        __builtin_prefetch(aRow0 + k0 + kcs, 0, 3);   // next A chunk -> WGP$
        __builtin_prefetch(aRow1 + k0 + kcs, 0, 3);
        for (int kk = 0; kk < kcs; kk += 32) {
            v16bf a0 = load_frag(aRow0 + k0 + kk + c8, aRow0 + k0 + kk + c8 + 16);
            v16bf a1 = load_frag(aRow1 + k0 + kk + c8, aRow1 + k0 + kk + c8 + 16);
#pragma unroll
            for (int j = 0; j < 4; ++j) {
                const unsigned short* bp = btile + (j * 16 + mrow) * kcs + kk + c8;
                v16bf b = load_frag(bp, bp + 16);
                acc[j * 2 + 0] = wmma_bf16(a0, b, acc[j * 2 + 0]);
                acc[j * 2 + 1] = wmma_bf16(a1, b, acc[j * 2 + 1]);
            }
        }
        __syncthreads();                         // reads done before next DMA
    }

    const int rb = (lane >> 4) * 8;
#pragma unroll
    for (int j = 0; j < 4; ++j) {
        const int col = col0 + j * 16 + mrow;
        const float bv = bias ? bias[col] : 0.0f;
#pragma unroll
        for (int r = 0; r < 2; ++r) {
            const int rowb = row0 + r * 16 + rb;
            v8f a = acc[j * 2 + r];
#pragma unroll
            for (int v = 0; v < 8; ++v) {
                const int rr = rowb + v;
                if (!dom || dom[rr] == domSel)
                    C[(size_t)rr * N + col] = a[v] + bv;
            }
        }
    }
}

// -------------------------------------------------------- LayerNorm+ReLU
__global__ void __launch_bounds__(128)
ln_relu_kernel(const float* __restrict__ X,
               const float* __restrict__ g, const float* __restrict__ be,
               const float* __restrict__ add,
               float* __restrict__ Yf, unsigned short* __restrict__ Yb, int N) {
    const int row = blockIdx.x;
    const float* x = X + (size_t)row * N;
    float s = 0.f, ss = 0.f;
    for (int i = threadIdx.x; i < N; i += 128) {
        float v = x[i]; s += v; ss += v * v;
    }
#pragma unroll
    for (int o = 16; o > 0; o >>= 1) {
        s  += __shfl_down(s,  o, 32);
        ss += __shfl_down(ss, o, 32);
    }
    __shared__ float sb[4], ssb[4];
    if ((threadIdx.x & 31) == 0) { sb[threadIdx.x >> 5] = s; ssb[threadIdx.x >> 5] = ss; }
    __syncthreads();
    const float tot  = sb[0] + sb[1] + sb[2] + sb[3];
    const float tot2 = ssb[0] + ssb[1] + ssb[2] + ssb[3];
    const float mean = tot / (float)N;
    const float var  = tot2 / (float)N - mean * mean;
    const float rstd = rsqrtf(var + 1e-5f);
    for (int i = threadIdx.x; i < N; i += 128) {
        float v = (x[i] - mean) * rstd * g[i] + be[i];
        v = v > 0.f ? v : 0.f;
        if (add) v += add[(size_t)row * N + i];
        if (Yf) Yf[(size_t)row * N + i] = v;
        if (Yb) Yb[(size_t)row * N + i] = f2bf(v);
    }
}

// -------------------------------------------------------------- gather
__global__ void gather_kernel(const float* __restrict__ tables,
                              const int* __restrict__ x,
                              unsigned short* __restrict__ embB) {
    int idx = blockIdx.x * blockDim.x + threadIdx.x;   // FF*BB
    if (idx >= FF * BB) return;
    int f = idx >> 14;
    int b = idx & (BB - 1);
    int id = x[(size_t)f * BB + b];
    const float* src = tables + ((size_t)f * VV + id) * EMBD;
    unsigned short* dst = embB + (size_t)b * IN + f * EMBD;
#pragma unroll
    for (int e = 0; e < EMBD; ++e) dst[e] = f2bf(src[e]);
}

// --------------------------------------------- weight convert + transpose
__global__ void cvt_t_kernel(const float* __restrict__ W,
                             const float* __restrict__ mul,
                             unsigned short* __restrict__ Wt, int K, int N) {
    int idx = blockIdx.x * blockDim.x + threadIdx.x;
    if (idx >= K * N) return;
    int n = idx % N, k = idx / N;
    float v = W[(size_t)k * N + n];
    if (mul) v *= mul[(size_t)k * N + n];
    Wt[(size_t)n * K + k] = f2bf(v);
}

__global__ void cvt_bf16_kernel(const float* __restrict__ X,
                                unsigned short* __restrict__ Y, int n) {
    int i = blockIdx.x * blockDim.x + threadIdx.x;
    if (i < n) Y[i] = f2bf(X[i]);
}

__global__ void combb_kernel(const float* __restrict__ slot_b,
                             const float* __restrict__ shared_b,
                             float* __restrict__ comb) {
    int i = blockIdx.x * blockDim.x + threadIdx.x;
    if (i < 3 * 1024) comb[i] = slot_b[i] + shared_b[i & 1023];
}

// ------------------------------------------------------- gate (softmax)
__global__ void __launch_bounds__(128)
gate_kernel(const float* __restrict__ h, const float* __restrict__ gW,
            const float* __restrict__ gb, float* __restrict__ gates) {
    int b = blockIdx.x * blockDim.x + threadIdx.x;
    int g = blockIdx.y;
    if (b >= BB) return;
    const float* hr = h + (size_t)b * IN;
    const float* w  = gW + (size_t)g * IN * 8;
    float acc[8];
#pragma unroll
    for (int e = 0; e < 8; ++e) acc[e] = gb[g * 8 + e];
    for (int k = 0; k < IN; ++k) {
        float hv = hr[k];
#pragma unroll
        for (int e = 0; e < 8; ++e) acc[e] += hv * w[k * 8 + e];
    }
    float mx = acc[0];
#pragma unroll
    for (int e = 1; e < 8; ++e) mx = fmaxf(mx, acc[e]);
    float sum = 0.f;
#pragma unroll
    for (int e = 0; e < 8; ++e) { acc[e] = __expf(acc[e] - mx); sum += acc[e]; }
    float inv = 1.f / sum;
#pragma unroll
    for (int e = 0; e < 8; ++e)
        gates[((size_t)g * BB + b) * 8 + e] = acc[e] * inv;
}

// ------------------------------------------ mix + balanced-expert fusion
__global__ void fuse_kernel(const float* __restrict__ gates,
                            const float* __restrict__ fea,   // (8,B,64)
                            const float* __restrict__ dexp,  // (3,B,64)
                            const float* __restrict__ texp,  // (2,B,64)
                            const float* __restrict__ wed_p, const float* __restrict__ wet_p,
                            const float* __restrict__ wbd_p, const float* __restrict__ wbt_p,
                            unsigned short* __restrict__ fusedB) {
    int idx = blockIdx.x * blockDim.x + threadIdx.x;        // 6*B*64
    if (idx >= 6 * BB * 64) return;
    int f = idx & 63;
    int bg = idx >> 6;
    int b = bg & (BB - 1);
    int g = bg >> 14;
    float m = 0.f;
#pragma unroll
    for (int e = 0; e < 8; ++e)
        m += gates[((size_t)g * BB + b) * 8 + e] * fea[((size_t)e * BB + b) * 64 + f];
    const float wbd = 1.f / (1.f + __expf(-wbd_p[0]));
    const float cd  = (1.f - wbd) * 0.5f;                    // /(D-1)
    const float d0 = dexp[((size_t)0 * BB + b) * 64 + f];
    const float d1 = dexp[((size_t)1 * BB + b) * 64 + f];
    const float d2 = dexp[((size_t)2 * BB + b) * 64 + f];
    const int   di = g % 3;
    const float ds = di == 0 ? d0 : (di == 1 ? d1 : d2);
    const float wdf = (wbd - cd) * ds + cd * (d0 + d1 + d2);
    const float wbt = 1.f / (1.f + __expf(-wbt_p[0]));
    const float ct  = (1.f - wbt);                           // /(T-1)
    const float t0 = texp[((size_t)0 * BB + b) * 64 + f];
    const float t1 = texp[((size_t)1 * BB + b) * 64 + f];
    const float ts = (g / 3) ? t1 : t0;
    const float wtf = (wbt - ct) * ts + ct * (t0 + t1);
    const float wed = 1.f / (1.f + __expf(-wed_p[0]));
    const float wet = 1.f / (1.f + __expf(-wet_p[0]));
    fusedB[idx] = f2bf(m + wed * wdf + wet * wtf);
}

// ----------------------------------------------------------- tower head
__global__ void tower_out_kernel(const float* __restrict__ towh,
                                 const float* __restrict__ W2,
                                 const float* __restrict__ b2,
                                 float* __restrict__ outs) {
    int b = blockIdx.x * blockDim.x + threadIdx.x;
    if (b >= BB) return;
    const float* xr = towh + (size_t)b * 64;
    float acc = b2[0];
#pragma unroll
    for (int f = 0; f < 64; ++f) acc += xr[f] * W2[f];
    outs[b] = 1.f / (1.f + __expf(-acc));
}

// ---------------------------------------------------- final dom select
__global__ void select_kernel(const float* __restrict__ outs,   // (6,B)
                              const int* __restrict__ dom,
                              float* __restrict__ out) {
    int idx = blockIdx.x * blockDim.x + threadIdx.x;   // B*2
    if (idx >= BB * 2) return;
    int b = idx >> 1, t = idx & 1;
    int d = dom[b];
    out[(size_t)b * 2 + t] = outs[(size_t)(t * 3 + d) * BB + b];
}

// =================================================================== host
extern "C" void kernel_launch(void* const* d_in, const int* in_sizes, int n_in,
                              void* d_out, int out_size, void* d_ws, size_t ws_size,
                              hipStream_t stream) {
    const int*   x        = (const int*)  d_in[0];
    const float* tables   = (const float*)d_in[1];
    const float* skip_W   = (const float*)d_in[2];
    const float* skip_b   = (const float*)d_in[3];
    const float* skip_g   = (const float*)d_in[4];
    const float* skip_be  = (const float*)d_in[5];
    const float* shared_w = (const float*)d_in[6];
    const float* shared_b = (const float*)d_in[7];
    const float* slot_w   = (const float*)d_in[8];
    const float* slot_b   = (const float*)d_in[9];
    const float* star_W   = (const float*)d_in[10];
    const float* star_b   = (const float*)d_in[11];
    const float* star_g   = (const float*)d_in[12];
    const float* star_be  = (const float*)d_in[13];
    const float* exp_W1   = (const float*)d_in[14];
    const float* exp_b1   = (const float*)d_in[15];
    const float* exp_g1   = (const float*)d_in[16];
    const float* exp_be1  = (const float*)d_in[17];
    const float* exp_W2   = (const float*)d_in[18];
    const float* exp_b2   = (const float*)d_in[19];
    const float* exp_g2   = (const float*)d_in[20];
    const float* exp_be2  = (const float*)d_in[21];
    const float* dexp_W1  = (const float*)d_in[22];
    const float* dexp_b1  = (const float*)d_in[23];
    const float* dexp_g1  = (const float*)d_in[24];
    const float* dexp_be1 = (const float*)d_in[25];
    const float* dexp_W2  = (const float*)d_in[26];
    const float* dexp_b2  = (const float*)d_in[27];
    const float* dexp_g2  = (const float*)d_in[28];
    const float* dexp_be2 = (const float*)d_in[29];
    const float* texp_W1  = (const float*)d_in[30];
    const float* texp_b1  = (const float*)d_in[31];
    const float* texp_g1  = (const float*)d_in[32];
    const float* texp_be1 = (const float*)d_in[33];
    const float* texp_W2  = (const float*)d_in[34];
    const float* texp_b2  = (const float*)d_in[35];
    const float* texp_g2  = (const float*)d_in[36];
    const float* texp_be2 = (const float*)d_in[37];
    const float* gate_W   = (const float*)d_in[38];
    const float* gate_b   = (const float*)d_in[39];
    const float* tw_W1    = (const float*)d_in[40];
    const float* tw_b1    = (const float*)d_in[41];
    const float* tw_g1    = (const float*)d_in[42];
    const float* tw_be1   = (const float*)d_in[43];
    const float* tw_W2    = (const float*)d_in[44];
    const float* tw_b2    = (const float*)d_in[45];
    const float* w_exp_d  = (const float*)d_in[46];
    const float* w_exp_t  = (const float*)d_in[47];
    const float* w_bal_d  = (const float*)d_in[48];
    const float* w_bal_t  = (const float*)d_in[49];
    (void)in_sizes; (void)n_in; (void)out_size; (void)ws_size;

    // ---- workspace bump allocator (256B aligned slots)
    char* ws = (char*)d_ws;
    size_t off = 0;
    auto alloc = [&](size_t bytes) -> char* {
        char* p = ws + off;
        off += (bytes + 255) & ~(size_t)255;
        return p;
    };
    typedef unsigned short us;
    us*    embB   = (us*)   alloc((size_t)BB * IN * 2);
    float* skipPre= (float*)alloc((size_t)BB * IN * 4);     // reused as hPre
    float* skipF  = (float*)alloc((size_t)BB * IN * 4);
    float* starPre= (float*)alloc((size_t)BB * 1024 * 4);
    us*    starB  = (us*)   alloc((size_t)BB * 1024 * 2);
    float* hF     = (float*)alloc((size_t)BB * IN * 4);
    us*    hB     = (us*)   alloc((size_t)BB * IN * 2);
    float* gates  = (float*)alloc((size_t)6 * BB * 8 * 4);
    float* e1Pre  = (float*)alloc((size_t)BB * 256 * 4);
    us*    e1B    = (us*)   alloc((size_t)BB * 256 * 2);
    float* e2Pre  = (float*)alloc((size_t)BB * 64 * 4);
    float* fea    = (float*)alloc((size_t)8 * BB * 64 * 4);
    float* dex    = (float*)alloc((size_t)3 * BB * 64 * 4);
    float* tex    = (float*)alloc((size_t)2 * BB * 64 * 4);
    us*    fusedB = (us*)   alloc((size_t)6 * BB * 64 * 2);
    float* towPre = (float*)alloc((size_t)BB * 64 * 4);
    float* towF   = (float*)alloc((size_t)BB * 64 * 4);
    float* outsB  = (float*)alloc((size_t)6 * BB * 4);
    float* combB  = (float*)alloc((size_t)3 * 1024 * 4);
    us* skipWt = (us*)alloc((size_t)512 * 512 * 2);
    us* effWt  = (us*)alloc((size_t)3 * 1024 * 512 * 2);
    us* starWt = (us*)alloc((size_t)512 * 1024 * 2);
    us* eW1t   = (us*)alloc((size_t)8 * 256 * 512 * 2);
    us* eW2t   = (us*)alloc((size_t)8 * 64 * 256 * 2);
    us* dW1t   = (us*)alloc((size_t)3 * 256 * 512 * 2);
    us* dW2t   = (us*)alloc((size_t)3 * 64 * 256 * 2);
    us* tW1t   = (us*)alloc((size_t)2 * 256 * 512 * 2);
    us* tW2t   = (us*)alloc((size_t)2 * 64 * 256 * 2);
    us* twW1t  = (us*)alloc((size_t)6 * 64 * 64 * 2);

    const int* dom = x + (size_t)(FF - 1) * BB;

    auto ct = [&](const float* W, const float* mul, us* Wt, int K, int N) {
        int n = K * N;
        cvt_t_kernel<<<(n + 255) / 256, 256, 0, stream>>>(W, mul, Wt, K, N);
    };
    auto gemm = [&](const us* A, const us* Wt, const float* bias, float* C,
                    int M, int N, int K, const int* dm, int ds) {
        gemm_bf16_kernel<<<dim3(M / 128, N / 64), 128, 0, stream>>>(
            A, Wt, bias, C, M, N, K, dm, ds);
    };
    auto lnrelu = [&](const float* X, const float* g, const float* be,
                      const float* add, float* Yf, us* Yb, int N) {
        ln_relu_kernel<<<BB, 128, 0, stream>>>(X, g, be, add, Yf, Yb, N);
    };

    // ---- 1. weight prep (bf16, transposed)
    ct(skip_W, nullptr, skipWt, IN, 512);
    for (int d = 0; d < 3; ++d)
        ct(slot_w + (size_t)d * IN * 1024, shared_w, effWt + (size_t)d * 1024 * IN, IN, 1024);
    ct(star_W, nullptr, starWt, 1024, 512);
    for (int e = 0; e < 8; ++e) {
        ct(exp_W1 + (size_t)e * 512 * 256, nullptr, eW1t + (size_t)e * 256 * 512, 512, 256);
        ct(exp_W2 + (size_t)e * 256 * 64,  nullptr, eW2t + (size_t)e * 64 * 256,  256, 64);
    }
    for (int e = 0; e < 3; ++e) {
        ct(dexp_W1 + (size_t)e * 512 * 256, nullptr, dW1t + (size_t)e * 256 * 512, 512, 256);
        ct(dexp_W2 + (size_t)e * 256 * 64,  nullptr, dW2t + (size_t)e * 64 * 256,  256, 64);
    }
    for (int e = 0; e < 2; ++e) {
        ct(texp_W1 + (size_t)e * 512 * 256, nullptr, tW1t + (size_t)e * 256 * 512, 512, 256);
        ct(texp_W2 + (size_t)e * 256 * 64,  nullptr, tW2t + (size_t)e * 64 * 256,  256, 64);
    }
    for (int g = 0; g < 6; ++g)
        ct(tw_W1 + (size_t)g * 64 * 64, nullptr, twW1t + (size_t)g * 64 * 64, 64, 64);
    combb_kernel<<<(3 * 1024 + 255) / 256, 256, 0, stream>>>(slot_b, shared_b, combB);

    // ---- 2. embedding gather -> bf16
    gather_kernel<<<(FF * BB + 255) / 256, 256, 0, stream>>>(tables, x, embB);

    // ---- 3. skip branch
    gemm(embB, skipWt, skip_b, skipPre, BB, 512, IN, nullptr, 0);
    lnrelu(skipPre, skip_g, skip_be, nullptr, skipF, nullptr, 512);

    // ---- 4. STAR per-domain GEMMs (each output row written by its domain)
    for (int d = 0; d < 3; ++d)
        gemm(embB, effWt + (size_t)d * 1024 * IN, combB + (size_t)d * 1024,
             starPre, BB, 1024, IN, dom, d);
    cvt_bf16_kernel<<<(BB * 1024 + 255) / 256, 256, 0, stream>>>(starPre, starB, BB * 1024);

    // ---- 5. h = block(star @ star_W) + skip
    gemm(starB, starWt, star_b, skipPre /*hPre*/, BB, 512, 1024, nullptr, 0);
    lnrelu(skipPre, star_g, star_be, skipF, hF, hB, 512);

    // ---- 6. gates = softmax(h @ gate_W + gate_b)
    gate_kernel<<<dim3((BB + 127) / 128, 6), 128, 0, stream>>>(hF, gate_W, gate_b, gates);

    // ---- 7. expert MLPs (8 shared, 3 domain, 2 task)
    auto expert = [&](const us* W1t, const float* b1, const float* g1, const float* be1,
                      const us* W2t, const float* b2, const float* g2, const float* be2,
                      float* outF) {
        gemm(hB, W1t, b1, e1Pre, BB, 256, 512, nullptr, 0);
        lnrelu(e1Pre, g1, be1, nullptr, nullptr, e1B, 256);
        gemm(e1B, W2t, b2, e2Pre, BB, 64, 256, nullptr, 0);
        lnrelu(e2Pre, g2, be2, nullptr, outF, nullptr, 64);
    };
    for (int e = 0; e < 8; ++e)
        expert(eW1t + (size_t)e * 256 * 512, exp_b1 + e * 256, exp_g1 + e * 256, exp_be1 + e * 256,
               eW2t + (size_t)e * 64 * 256,  exp_b2 + e * 64,  exp_g2 + e * 64,  exp_be2 + e * 64,
               fea + (size_t)e * BB * 64);
    for (int e = 0; e < 3; ++e)
        expert(dW1t + (size_t)e * 256 * 512, dexp_b1 + e * 256, dexp_g1 + e * 256, dexp_be1 + e * 256,
               dW2t + (size_t)e * 64 * 256,  dexp_b2 + e * 64,  dexp_g2 + e * 64,  dexp_be2 + e * 64,
               dex + (size_t)e * BB * 64);
    for (int e = 0; e < 2; ++e)
        expert(tW1t + (size_t)e * 256 * 512, texp_b1 + e * 256, texp_g1 + e * 256, texp_be1 + e * 256,
               tW2t + (size_t)e * 64 * 256,  texp_b2 + e * 64,  texp_g2 + e * 64,  texp_be2 + e * 64,
               tex + (size_t)e * BB * 64);

    // ---- 8. gate mix + balanced fusion -> bf16 tower input
    fuse_kernel<<<(6 * BB * 64 + 255) / 256, 256, 0, stream>>>(
        gates, fea, dex, tex, w_exp_d, w_exp_t, w_bal_d, w_bal_t, fusedB);

    // ---- 9. towers
    for (int g = 0; g < 6; ++g) {
        gemm(fusedB + (size_t)g * BB * 64, twW1t + (size_t)g * 64 * 64,
             tw_b1 + g * 64, towPre, BB, 64, 64, nullptr, 0);
        lnrelu(towPre, tw_g1 + g * 64, tw_be1 + g * 64, nullptr, towF, nullptr, 64);
        tower_out_kernel<<<(BB + 255) / 256, 256, 0, stream>>>(
            towF, tw_W2 + (size_t)g * 64, tw_b2 + g, outsB + (size_t)g * BB);
    }

    // ---- 10. domain select -> (B, T)
    select_kernel<<<(BB * 2 + 255) / 256, 256, 0, stream>>>(outsB, dom, (float*)d_out);
}
